// DKNET_33079838113962
// MI455X (gfx1250) — matrix operand order repaired and checked by reading, not compiled, not run on hardware
//
#include <hip/hip_runtime.h>
#include <hip/hip_bf16.h>

// ---------------------------------------------------------------------------
// DKNET pansharpening net on gfx1250.
//   * All ten 1x1 convs run through V_WMMA_F32_16X16X32_BF16 (K=32 matches the
//     32-channel layers exactly; cin=8/9 layers are zero-padded in K).
//   * Each wave amortizes one A-fragment load over 4 pixel tiles -> 8 WMMAs
//     per wave, statically unrolled.
//   * Conv/DDF/depthwise kernels are fully specialized (template CIN/COUT/
//     ACT/ACC, HW=65536 constexpr) so channel offsets fold into the 24-bit
//     VMEM instruction offsets; the store epilogue has a single divergence
//     region (slot validity is static per lane half).
//   * Weight matrices are repacked per launch into the documented CDNA5
//     16-bit A-fragment layout (16x32 MxK, lane/VGPR striping per ISA 7.12.2).
//   * Residual / lms skip adds are fused into the DDF / conv store paths.
// ---------------------------------------------------------------------------

typedef __attribute__((ext_vector_type(16))) __bf16 v16bf;
typedef __attribute__((ext_vector_type(8)))  float  v8f;

__device__ __forceinline__ unsigned short f2bf(float f) {
    unsigned int u = __float_as_uint(f);
    unsigned int r = u + 0x7FFFu + ((u >> 16) & 1u);   // round-to-nearest-even
    return (unsigned short)(r >> 16);
}

// Fast sigmoid: v_exp_f32 + v_rcp_f32 (avoid the correctly-rounded divide).
__device__ __forceinline__ float sigmoidf(float x) {
    return __builtin_amdgcn_rcpf(1.0f + __expf(-x));
}

template <int ACT>
__device__ __forceinline__ float act_f(float v) {
    if constexpr (ACT == 1) return fmaxf(v, 0.0f);
    else if constexpr (ACT == 2) return sigmoidf(v);
    else return v;
}

// ---------------------------------------------------------------------------
// Pack W (cout x cin, row-major fp32) into two 16x32 bf16 A-fragments.
// Layout (ISA 7.12.2, 16-bit A 16x32): lane L: M = L&15 (+16 for tile 1),
// half = L>>4; VGPR v holds K = (v>>2)*16 + half*8 + (v&3)*2 and K+1,
// low element in [15:0]. 64 threads = 2 tiles x 32 lanes; 8 dwords per lane.
// ---------------------------------------------------------------------------
__global__ void pack_a_kernel(const float* __restrict__ w, int cout, int cin,
                              unsigned int* __restrict__ dst) {
    int t    = threadIdx.x;      // 0..63
    int tile = t >> 5;
    int lane = t & 31;
    int m    = tile * 16 + (lane & 15);
    int hl   = lane >> 4;
#pragma unroll
    for (int v = 0; v < 8; ++v) {
        int k0 = (v >> 2) * 16 + hl * 8 + (v & 3) * 2;
        unsigned short e0 = (m < cout && k0     < cin) ? f2bf(w[m * cin + k0    ]) : (unsigned short)0;
        unsigned short e1 = (m < cout && k0 + 1 < cin) ? f2bf(w[m * cin + k0 + 1]) : (unsigned short)0;
        dst[t * 8 + v] = (unsigned int)e0 | ((unsigned int)e1 << 16);
    }
}

// ---------------------------------------------------------------------------
// 1x1 conv as WMMA GEMM: out[b,m,p] = sum_k W[m,k]*x[b,k,p] + bias[m].
// One wave handles 4 consecutive 16-pixel tiles (64 pixels), reusing the
// A fragments across tiles.  B fragment (32x16 KxN bf16) is built from fp32
// activations on the fly (ISA B layout: lane L -> N=L&15, half=L>>4;
// VGPR r holds K = (r>>2)*16 + half*8 + (r&3)*2, K+1).
// ACT: 0 none, 1 relu, 2 sigmoid.  ACC: out += conv result first.
// ---------------------------------------------------------------------------
template <int CIN, int COUT, int ACT, bool ACC>
__global__ void __launch_bounds__(256)
conv1x1_wmma_kernel(const float* __restrict__ x,
                    const unsigned int* __restrict__ apack,
                    const float* __restrict__ bias,
                    float* __restrict__ out) {
    constexpr int  HW   = 65536;           // 256*256, fixed problem size
    constexpr bool TWO  = (COUT > 16);
    // Tile-0 store-slot structure (slot v -> row m = hl*8 + v):
    //   v <  VFULL : valid for both lane halves  -> unconditional store
    //   v in [VFULL, VHALF) : valid only for hl==0 -> one divergence region
    constexpr int VFULL = (COUT >= 16) ? 8 : ((COUT > 8) ? COUT - 8 : 0);
    constexpr int VHALF = (COUT < 8) ? COUT : 8;

    const int lane = threadIdx.x & 31;
    const int wave = blockIdx.x * (blockDim.x >> 5) + (threadIdx.x >> 5);
    const int b   = wave >> 10;                 // HW/64 = 1024 groups per image
    const int pg  = (wave & 1023) << 6;         // first pixel of 64-px group
    const int n   = lane & 15;
    const int hl  = lane >> 4;

    const float* xb = x   + (size_t)b * (size_t)CIN  * HW;
    float*       ob = out + (size_t)b * (size_t)COUT * HW;

    // A fragments: 8 dwords per lane, lane-major (2 tiles x 32 lanes x 8).
    const uint4* ap = (const uint4*)apack;
    union { uint4 q[2]; v16bf v; } a0, a1;
    a0.q[0] = ap[lane * 2 + 0];
    a0.q[1] = ap[lane * 2 + 1];
    if constexpr (TWO) {
        a1.q[0] = ap[(32 + lane) * 2 + 0];
        a1.q[1] = ap[(32 + lane) * 2 + 1];
    }

    // Per-lane biases (hoisted out of the tile loop; same divergence shape
    // as the stores).
    const int m0 = hl * 8;
    float bias0[8] = {}, bias1[8] = {};
#pragma unroll
    for (int v = 0; v < VFULL; ++v) bias0[v] = bias[m0 + v];
    if constexpr (VHALF > VFULL) {
        if (hl == 0) {
#pragma unroll
            for (int v = VFULL; v < VHALF; ++v) bias0[v] = bias[v];
        }
    }
    if constexpr (TWO) {
#pragma unroll
        for (int v = 0; v < 8; ++v) bias1[v] = bias[16 + m0 + v];
    }

#pragma unroll
    for (int t = 0; t < 4; ++t) {
        const int p = pg + t * 16 + n;
        const float* xp = xb + p;

        // Build B fragment (activations -> bf16, RNE). Channel offsets are
        // compile-time -> fold into VMEM instruction offsets.
        union { unsigned int u[8]; v16bf v; } bb;
#pragma unroll
        for (int r = 0; r < 8; ++r) {
            const int k0 = (r >> 2) * 16 + hl * 8 + (r & 3) * 2;
            float f0 = (k0     < CIN) ? xp[k0 * HW]       : 0.0f;
            float f1 = (k0 + 1 < CIN) ? xp[(k0 + 1) * HW] : 0.0f;
            bb.u[r] = (unsigned int)f2bf(f0) | ((unsigned int)f2bf(f1) << 16);
        }

        v8f c0 = {};
        c0 = __builtin_amdgcn_wmma_f32_16x16x32_bf16(false, a0.v, false, bb.v,
                                                     (short)0, c0, false, false);
        v8f c1 = {};
        if constexpr (TWO)
            c1 = __builtin_amdgcn_wmma_f32_16x16x32_bf16(false, a1.v, false, bb.v,
                                                         (short)0, c1, false, false);

        // Store epilogue: one base pointer per region, compile-time v*HW
        // offsets (<= 1.75MB, inside the signed 24-bit IOFFSET).
        float* op  = ob + p;
        float* oph = op + m0 * HW;
#pragma unroll
        for (int v = 0; v < VFULL; ++v) {
            float val = c0[v] + bias0[v];
            if constexpr (ACC) val += oph[v * HW];
            oph[v * HW] = act_f<ACT>(val);
        }
        if constexpr (VHALF > VFULL) {
            if (hl == 0) {
#pragma unroll
                for (int v = VFULL; v < VHALF; ++v) {
                    float val = c0[v] + bias0[v];
                    if constexpr (ACC) val += op[v * HW];
                    op[v * HW] = act_f<ACT>(val);
                }
            }
        }
        if constexpr (TWO) {
            float* opt = op + (16 + m0) * HW;
#pragma unroll
            for (int v = 0; v < 8; ++v) {
                float val = c1[v] + bias1[v];
                if constexpr (ACC) val += opt[v * HW];
                opt[v * HW] = act_f<ACT>(val);
            }
        }
    }
}

// ---------------------------------------------------------------------------
// Global average pool: one block per (b,c) plane.
// ---------------------------------------------------------------------------
__global__ void gap_kernel(const float* __restrict__ x, float* __restrict__ mean) {
    constexpr int HW = 65536;
    const float* p = x + (size_t)blockIdx.x * HW;
    float s = 0.0f;
    for (int i = threadIdx.x; i < HW; i += 256) s += p[i];
    __shared__ float red[256];
    red[threadIdx.x] = s;
    __syncthreads();
    for (int w = 128; w > 0; w >>= 1) {
        if ((int)threadIdx.x < w) red[threadIdx.x] += red[threadIdx.x + w];
        __syncthreads();
    }
    if (threadIdx.x == 0) mean[blockIdx.x] = red[0] * (1.0f / (float)HW);
}

// ---------------------------------------------------------------------------
// DDF channel branch: cf[b, c*9+k] = sigmoid(cw2 . relu(cw1 . mean + cb1) + cb2)
// One block per batch image; mid <= 6, C*9 <= 288.
// ---------------------------------------------------------------------------
__global__ void cf_kernel(const float* __restrict__ mean,
                          const float* __restrict__ cw1, const float* __restrict__ cb1,
                          const float* __restrict__ cw2, const float* __restrict__ cb2,
                          float* __restrict__ cf, int C, int mid) {
    int b = blockIdx.x;
    int t = threadIdx.x;
    __shared__ float h[32];
    if (t < mid) {
        float s = cb1[t];
        for (int ci = 0; ci < C; ++ci) s += cw1[t * C + ci] * mean[b * C + ci];
        h[t] = fmaxf(s, 0.0f);
    }
    __syncthreads();
    if (t < C * 9) {
        float s = cb2[t];
        for (int m = 0; m < mid; ++m) s += cw2[t * mid + m] * h[m];
        cf[b * C * 9 + t] = sigmoidf(s);
    }
}

// ---------------------------------------------------------------------------
// DDF tap-sum: out[b,c,p] = sum_k x[b,c,nbr_k(p)] * sf[b,k,p] * cf[b,c,k]
// (+ optional relu, + optional skip).  16x16 pixel tile per block with an
// 18x18 LDS halo per channel; sf (already sigmoided) kept in registers.
// ---------------------------------------------------------------------------
template <int C, bool RELU, bool SKIP>
__global__ void __launch_bounds__(256)
ddf_kernel(const float* __restrict__ in, const float* __restrict__ sf,
           const float* __restrict__ cf, float* __restrict__ out,
           const float* __restrict__ skip) {
    constexpr int H = 256, W = 256, HW = H * W;
    const int b  = blockIdx.z;
    const int tx = threadIdx.x, ty = threadIdx.y;
    const int gx = blockIdx.x * 16 + tx;
    const int gy = blockIdx.y * 16 + ty;
    const int p  = gy * W + gx;
    const int lt = ty * 16 + tx;

    float s[9];
#pragma unroll
    for (int k = 0; k < 9; ++k) s[k] = sf[((size_t)b * 9 + k) * HW + p];

    __shared__ float tile[18][19];
    __shared__ float cfs[C * 9];
    for (int i = lt; i < C * 9; i += 256) cfs[i] = cf[b * C * 9 + i];

    for (int c = 0; c < C; ++c) {
        __syncthreads();   // covers cfs load (iter 0) and prior tile reads
        const float* ip = in + ((size_t)b * C + c) * HW;
        for (int i = lt; i < 18 * 18; i += 256) {
            int yy = i / 18, xx = i - yy * 18;
            int sy = blockIdx.y * 16 + yy - 1;
            int sx = blockIdx.x * 16 + xx - 1;
            tile[yy][xx] = (sy >= 0 && sy < H && sx >= 0 && sx < W) ? ip[sy * W + sx] : 0.0f;
        }
        __syncthreads();
        float acc = 0.0f;
#pragma unroll
        for (int ki = 0; ki < 3; ++ki)
#pragma unroll
            for (int kj = 0; kj < 3; ++kj) {
                int k = ki * 3 + kj;
                acc = fmaf(tile[ty + ki][tx + kj], s[k] * cfs[c * 9 + k], acc);
            }
        float r = RELU ? fmaxf(acc, 0.0f) : acc;
        size_t oidx = ((size_t)b * C + c) * HW + p;
        if (SKIP) r += skip[oidx];
        out[oidx] = r;
    }
}

// ---------------------------------------------------------------------------
// Depthwise 3x3 (SAME, stride 1) + bias.  One (b,c) plane per blockIdx.z.
// ---------------------------------------------------------------------------
template <int C>
__global__ void __launch_bounds__(256)
dw3x3_kernel(const float* __restrict__ in, const float* __restrict__ w,
             const float* __restrict__ bias, float* __restrict__ out) {
    constexpr int H = 256, W = 256, HW = H * W;
    const int bc = blockIdx.z;
    const int c  = bc & (C - 1);          // C is a power of two (8 or 32)
    const int tx = threadIdx.x, ty = threadIdx.y;
    const int gx = blockIdx.x * 16 + tx;
    const int gy = blockIdx.y * 16 + ty;
    const int lt = ty * 16 + tx;

    __shared__ float tile[18][19];
    const float* ip = in + (size_t)bc * HW;
    for (int i = lt; i < 18 * 18; i += 256) {
        int yy = i / 18, xx = i - yy * 18;
        int sy = blockIdx.y * 16 + yy - 1;
        int sx = blockIdx.x * 16 + xx - 1;
        tile[yy][xx] = (sy >= 0 && sy < H && sx >= 0 && sx < W) ? ip[sy * W + sx] : 0.0f;
    }
    __syncthreads();
    float acc = bias[c];
#pragma unroll
    for (int ki = 0; ki < 3; ++ki)
#pragma unroll
        for (int kj = 0; kj < 3; ++kj)
            acc = fmaf(tile[ty + ki][tx + kj], w[c * 9 + ki * 3 + kj], acc);
    out[(size_t)bc * HW + gy * W + gx] = acc;
}

// ---------------------------------------------------------------------------
// Concat pan(1ch) + lms(8ch) -> x9.
// ---------------------------------------------------------------------------
__global__ void concat_kernel(const float* __restrict__ pan,
                              const float* __restrict__ lms,
                              float* __restrict__ x9) {
    constexpr int HW = 65536;
    size_t idx = (size_t)blockIdx.x * blockDim.x + threadIdx.x;
    int b  = (int)(idx / (9 * (size_t)HW));
    int r  = (int)(idx - (size_t)b * 9 * HW);
    int c  = r / HW;
    int hw = r - c * HW;
    x9[idx] = (c == 0) ? pan[(size_t)b * HW + hw]
                       : lms[((size_t)b * 8 + (c - 1)) * HW + hw];
}

// ---------------------------------------------------------------------------
// Host side.
// ---------------------------------------------------------------------------
extern "C" void kernel_launch(void* const* d_in, const int* in_sizes, int n_in,
                              void* d_out, int out_size, void* d_ws, size_t ws_size,
                              hipStream_t stream) {
    (void)in_sizes; (void)n_in; (void)out_size; (void)ws_size;

    const int NB = 4, H = 256, W = 256;
    const int HW = H * W;

    // d_in order: pan, lms, then params flattened jax-tree style (sorted keys
    // at each level): head_bs{dw_b,dw_w,pw_b,pw_w,pwb_b,pwb_w},
    // head_ddf{cb1,cb2,cw1,cw2,sb,sw}, rb1_c1, rb1_c2, rb2_c1, rb2_c2 (ddf),
    // tail_bs (bs), tail_ddf (ddf).
    const float* pan = (const float*)d_in[0];
    const float* lms = (const float*)d_in[1];
    int i = 2;
    // head_bs
    const float* hb_dwb  = (const float*)d_in[i++]; const float* hb_dww  = (const float*)d_in[i++];
    const float* hb_pwb_ = (const float*)d_in[i++]; const float* hb_pww  = (const float*)d_in[i++];
    const float* hb_pbb  = (const float*)d_in[i++]; const float* hb_pbw  = (const float*)d_in[i++];
    // ddf layers: head, rb1_c1, rb1_c2, rb2_c1, rb2_c2  (tail_ddf after tail_bs)
    const float *dd_cb1[6], *dd_cb2[6], *dd_cw1[6], *dd_cw2[6], *dd_sb[6], *dd_sw[6];
    for (int l = 0; l < 5; ++l) {
        dd_cb1[l] = (const float*)d_in[i++]; dd_cb2[l] = (const float*)d_in[i++];
        dd_cw1[l] = (const float*)d_in[i++]; dd_cw2[l] = (const float*)d_in[i++];
        dd_sb[l]  = (const float*)d_in[i++]; dd_sw[l]  = (const float*)d_in[i++];
    }
    // tail_bs
    const float* tb_dwb  = (const float*)d_in[i++]; const float* tb_dww  = (const float*)d_in[i++];
    const float* tb_pwb_ = (const float*)d_in[i++]; const float* tb_pww  = (const float*)d_in[i++];
    const float* tb_pbb  = (const float*)d_in[i++]; const float* tb_pbw  = (const float*)d_in[i++];
    // tail_ddf
    dd_cb1[5] = (const float*)d_in[i++]; dd_cb2[5] = (const float*)d_in[i++];
    dd_cw1[5] = (const float*)d_in[i++]; dd_cw2[5] = (const float*)d_in[i++];
    dd_sb[5]  = (const float*)d_in[i++]; dd_sw[5]  = (const float*)d_in[i++];

    // Workspace carve-up (floats).  Everything fits in L2 (192 MB).
    float* ws = (float*)d_ws;
    size_t off = 0;
    float* x9   = ws + off; off += (size_t)NB * 9  * HW;   // concat input
    float* d9   = ws + off; off += (size_t)NB * 9  * HW;   // head ddf out
    float* sfb  = ws + off; off += (size_t)NB * 9  * HW;   // per-pixel spatial gate
    float* t32a = ws + off; off += (size_t)NB * 32 * HW;
    float* t32b = ws + off; off += (size_t)NB * 32 * HW;
    float* t32c = ws + off; off += (size_t)NB * 32 * HW;
    float* t8a  = ws + off; off += (size_t)NB * 8  * HW;
    float* t8b  = ws + off; off += (size_t)NB * 8  * HW;
    float* mean = ws + off; off += 128;                    // <= B*32
    float* cfb  = ws + off; off += 1152;                   // <= B*32*9
    unsigned int* apack = (unsigned int*)(ws + off);       // 10 layers x 512 dwords

    auto AP = [&](int layer) { return apack + layer * 512; };

    // --- Pack all WMMA A-fragments (weights -> bf16 fragments) ---
    // layers 0..5: ddf sw (head, rb1_c1, rb1_c2, rb2_c1, rb2_c2, tail)
    const int ddC[6] = {9, 32, 32, 32, 32, 8};
    for (int l = 0; l < 6; ++l)
        pack_a_kernel<<<1, 64, 0, stream>>>(dd_sw[l], 9, ddC[l], AP(l));
    pack_a_kernel<<<1, 64, 0, stream>>>(hb_pww, 32, 9,  AP(6));  // head pw
    pack_a_kernel<<<1, 64, 0, stream>>>(hb_pbw, 32, 9,  AP(7));  // head pwb
    pack_a_kernel<<<1, 64, 0, stream>>>(tb_pww,  8, 32, AP(8));  // tail pw
    pack_a_kernel<<<1, 64, 0, stream>>>(tb_pbw,  8, 32, AP(9));  // tail pwb

    const int convBlocks = (NB * HW / 64) / 8;   // 8 waves x 64 px per block
    dim3 tileGrid(W / 16, H / 16, NB), tileBlk(16, 16);

    auto gap_cf = [&](int l, const float* in, int C, int mid) {
        gap_kernel<<<NB * C, 256, 0, stream>>>(in, mean);
        cf_kernel<<<NB, 288, 0, stream>>>(mean, dd_cw1[l], dd_cb1[l],
                                          dd_cw2[l], dd_cb2[l], cfb, C, mid);
    };

    // --- Network ---
    concat_kernel<<<(NB * 9 * HW) / 256, 256, 0, stream>>>(pan, lms, x9);

    // head: DDF(9) -> BSConv(9->32) -> ReLU
    gap_cf(0, x9, 9, 1);
    conv1x1_wmma_kernel<9, 9, 2, false><<<convBlocks, 256, 0, stream>>>(x9, AP(0), dd_sb[0], sfb);
    ddf_kernel<9, false, false><<<tileGrid, tileBlk, 0, stream>>>(x9, sfb, cfb, d9, nullptr);

    conv1x1_wmma_kernel<9, 32, 0, false><<<convBlocks, 256, 0, stream>>>(d9, AP(6), hb_pwb_, t32a);
    dw3x3_kernel<32><<<dim3(W / 16, H / 16, NB * 32), tileBlk, 0, stream>>>(t32a, hb_dww, hb_dwb, t32b);
    conv1x1_wmma_kernel<9, 32, 1, true><<<convBlocks, 256, 0, stream>>>(d9, AP(7), hb_pbb, t32b);

    // rb1: x = x + ddf(relu(ddf(x)))
    gap_cf(1, t32b, 32, 6);
    conv1x1_wmma_kernel<32, 9, 2, false><<<convBlocks, 256, 0, stream>>>(t32b, AP(1), dd_sb[1], sfb);
    ddf_kernel<32, true, false><<<tileGrid, tileBlk, 0, stream>>>(t32b, sfb, cfb, t32a, nullptr);

    gap_cf(2, t32a, 32, 6);
    conv1x1_wmma_kernel<32, 9, 2, false><<<convBlocks, 256, 0, stream>>>(t32a, AP(2), dd_sb[2], sfb);
    ddf_kernel<32, false, true><<<tileGrid, tileBlk, 0, stream>>>(t32a, sfb, cfb, t32c, t32b);

    // rb2
    gap_cf(3, t32c, 32, 6);
    conv1x1_wmma_kernel<32, 9, 2, false><<<convBlocks, 256, 0, stream>>>(t32c, AP(3), dd_sb[3], sfb);
    ddf_kernel<32, true, false><<<tileGrid, tileBlk, 0, stream>>>(t32c, sfb, cfb, t32a, nullptr);

    gap_cf(4, t32a, 32, 6);
    conv1x1_wmma_kernel<32, 9, 2, false><<<convBlocks, 256, 0, stream>>>(t32a, AP(4), dd_sb[4], sfb);
    ddf_kernel<32, false, true><<<tileGrid, tileBlk, 0, stream>>>(t32a, sfb, cfb, t32b, t32c);

    // tail: BSConv(32->8) -> DDF(8), output = lms + result
    conv1x1_wmma_kernel<32, 8, 0, false><<<convBlocks, 256, 0, stream>>>(t32b, AP(8), tb_pwb_, t8a);
    dw3x3_kernel<8><<<dim3(W / 16, H / 16, NB * 8), tileBlk, 0, stream>>>(t8a, tb_dww, tb_dwb, t8b);
    conv1x1_wmma_kernel<32, 8, 0, true><<<convBlocks, 256, 0, stream>>>(t32b, AP(9), tb_pbb, t8b);

    gap_cf(5, t8b, 8, 1);
    conv1x1_wmma_kernel<8, 9, 2, false><<<convBlocks, 256, 0, stream>>>(t8b, AP(5), dd_sb[5], sfb);
    ddf_kernel<8, false, true><<<tileGrid, tileBlk, 0, stream>>>(t8b, sfb, cfb, (float*)d_out, lms);
}
